// Atten_Head_4234837753948
// MI455X (gfx1250) — matrix operand order repaired and checked by reading, not compiled, not run on hardware
//
#include <hip/hip_runtime.h>

// MI455X / gfx1250, wave32. bf16 WMMA (16x16x32, f32 accum) for QKV GEMMs and
// flash attention. Workspace: Q,K,V as bf16 [4][2048][1024] each = 50.3 MB.
// K/Q tiles staged with GLOBAL_LOAD_ASYNC_TO_LDS (ASYNCcnt); V staged through
// VGPRs because it is transposed into LDS (B-operand layout).

typedef __attribute__((ext_vector_type(16))) __bf16 v16bf;
typedef __attribute__((ext_vector_type(8)))  __bf16 v8bf;
typedef __attribute__((ext_vector_type(8)))  float  v8f;

union V16  { v16bf v; v8bf h[2]; };
union BF8x { uint4 u; __bf16 e[8]; };

static __device__ __forceinline__ v8f wmma_bf16(const V16& a, const V16& b, v8f c) {
  return __builtin_amdgcn_wmma_f32_16x16x32_bf16(false, a.v, false, b.v,
                                                 (short)0, c, false, false);
}

// Async global->LDS copy, 16B per lane. ldsOff = per-lane LDS byte address,
// byteOff = per-lane byte offset from uniform 64-bit base (GVS mode).
static __device__ __forceinline__ void async_g2lds_b128(unsigned ldsOff,
                                                        const void* gbase,
                                                        unsigned byteOff) {
  asm volatile("global_load_async_to_lds_b128 %0, %1, %2"
               :: "v"(ldsOff), "v"(byteOff), "s"(gbase)
               : "memory");
}
static __device__ __forceinline__ void wait_asynccnt0() {
  asm volatile("s_wait_asynccnt 0x0" ::: "memory");
}
// Low 32 bits of a generic pointer into LDS aperture = LDS byte address.
static __device__ __forceinline__ unsigned lds_addr(const void* p) {
  return (unsigned)(unsigned long long)p;
}

// ---------------------------------------------------------------------------
// Kernel 1: Y[bf16] = (X[f32] @ W[f32]) * scale    M=8192, K=N=1024
// 256 threads = 8 waves; block tile 128x128, K-step 32.
// LDS: Xs [128][K=32] row-major (A operand), Ws [N=128][K=32] (B operand).
// f32->bf16 conversion happens on the staging path, so no async copy here.
// ---------------------------------------------------------------------------
#define GP 40  // padded K pitch (elems): 16B-aligned chunks, bank-spread rows

__global__ __launch_bounds__(256) void qkv_proj_bf16(
    const float* __restrict__ X, const float* __restrict__ W,
    __bf16* __restrict__ Y, float scale)
{
  const int K = 1024, N = 1024;
  __shared__ __bf16 Xs[128 * GP];
  __shared__ __bf16 Ws[128 * GP];

  const int tid = threadIdx.x, lane = tid & 31, wave = tid >> 5;
  const int hi = lane >> 4, lo = lane & 15;
  const int mBlock = blockIdx.y * 128, nBlock = blockIdx.x * 128;
  const int waveM = wave >> 1, waveN = wave & 1;

  const v8f vzero = {0.f,0.f,0.f,0.f,0.f,0.f,0.f,0.f};
  v8f acc[2][4];
  for (int i = 0; i < 2; ++i)
    for (int j = 0; j < 4; ++j) acc[i][j] = vzero;

  for (int kk = 0; kk < K; kk += 32) {
    // stage X tile 128x32 (f32 -> bf16)
    #pragma unroll
    for (int i = 0; i < 4; ++i) {
      int idx = tid + i * 256;
      int r = idx >> 3, c = (idx & 7) * 4;
      float4 f = *(const float4*)(X + (size_t)(mBlock + r) * K + kk + c);
      __bf16* d = &Xs[r * GP + c];
      d[0] = (__bf16)f.x; d[1] = (__bf16)f.y; d[2] = (__bf16)f.z; d[3] = (__bf16)f.w;
    }
    // stage W tile 32x128, transposed to [N][K] (B-operand friendly)
    #pragma unroll
    for (int i = 0; i < 4; ++i) {
      int idx = tid + i * 256;
      int kr = idx >> 5, n0 = (idx & 31) * 4;
      float4 f = *(const float4*)(W + (size_t)(kk + kr) * N + nBlock + n0);
      Ws[(n0 + 0) * GP + kr] = (__bf16)f.x;
      Ws[(n0 + 1) * GP + kr] = (__bf16)f.y;
      Ws[(n0 + 2) * GP + kr] = (__bf16)f.z;
      Ws[(n0 + 3) * GP + kr] = (__bf16)f.w;
    }
    __syncthreads();

    // A operand: lane(l<16) holds K 0..7,16..23 ; lane(l>=16) holds 8..15,24..31
    V16 a[2];
    #pragma unroll
    for (int mt = 0; mt < 2; ++mt) {
      const __bf16* ab = &Xs[(waveM * 32 + mt * 16 + lo) * GP + hi * 8];
      a[mt].h[0] = *(const v8bf*)(ab);
      a[mt].h[1] = *(const v8bf*)(ab + 16);
    }
    // B operand: lane(l<16) holds K 0..7,16..23 ; lane(l>=16) holds 8..15,24..31 of column N = lo
    #pragma unroll
    for (int nt = 0; nt < 4; ++nt) {
      const __bf16* bb = &Ws[(waveN * 64 + nt * 16 + lo) * GP + hi * 8];
      V16 b;
      b.h[0] = *(const v8bf*)(bb);
      b.h[1] = *(const v8bf*)(bb + 16);
      acc[0][nt] = wmma_bf16(a[0], b, acc[0][nt]);
      acc[1][nt] = wmma_bf16(a[1], b, acc[1][nt]);
    }
    __syncthreads();
  }

  // C layout: VGPR r -> row r (lanes 0-15) / row r+8 (lanes 16-31), col = lane&15
  for (int mt = 0; mt < 2; ++mt)
    for (int nt = 0; nt < 4; ++nt)
      #pragma unroll
      for (int r = 0; r < 8; ++r) {
        int row = mBlock + waveM * 32 + mt * 16 + hi * 8 + r;
        int col = nBlock + waveN * 64 + nt * 16 + lo;
        Y[(size_t)row * N + col] = (__bf16)(acc[mt][nt][r] * scale);
      }
}

// ---------------------------------------------------------------------------
// Kernel 2: causal flash attention. Q pre-scaled by 1/sqrt(1024).
// Block = 256 threads (8 waves) handles one batch x 32 query rows.
// LDS: Qs [32][1032], Ks [32][1032], Vt [1024][40] (V transposed -> B operand),
//      Sb partial scores (2 d-halves), Pb probs (A operand), row stats.
// ---------------------------------------------------------------------------
#define QP   1032
#define KPCH 1032
#define VPCH 40
#define PPCH 40

__global__ __launch_bounds__(256) void flash_attn_causal(
    const __bf16* __restrict__ Qg, const __bf16* __restrict__ Kg,
    const __bf16* __restrict__ Vg, float* __restrict__ O)
{
  const int S = 2048, D = 1024;
  extern __shared__ char smem[];
  __bf16* Qs   = (__bf16*)smem;               // 32*QP
  __bf16* Ks   = Qs + 32 * QP;                // 32*KPCH
  __bf16* Vt   = Ks + 32 * KPCH;              // 1024*VPCH
  float*  Sb   = (float*)(Vt + 1024 * VPCH);  // 2 * 32 * 32
  __bf16* Pb   = (__bf16*)(Sb + 2 * 32 * 32); // 32*PPCH
  float*  rowM = (float*)(Pb + 32 * PPCH);
  float*  rowL = rowM + 32;
  float*  rowC = rowL + 32;

  const int tid = threadIdx.x, lane = tid & 31, wave = tid >> 5;
  const int hi = lane >> 4, lo = lane & 15;
  const int batch = blockIdx.y;
  const int qBase = blockIdx.x * 32;
  const size_t bOff = (size_t)batch * S * D;

  if (tid < 32) { rowM[tid] = -__builtin_inff(); rowL[tid] = 0.f; rowC[tid] = 0.f; }

  // stage Q block 32 x 1024 asynchronously (bf16, 16B per lane per issue)
  {
    const void* gq = (const void*)(Qg + bOff + (size_t)qBase * D);
    const unsigned qsBase = lds_addr(Qs);
    for (int i = tid; i < 32 * 128; i += 256) {
      int r = i >> 7, c = (i & 127) * 8;
      async_g2lds_b128(qsBase + (unsigned)(r * QP + c) * 2u, gq,
                       (unsigned)(r * D + c) * 2u);
    }
  }

  const int mb = wave >> 2;       // 0..1 : 16-row group for PV/output
  const int nb = wave & 3;        // 0..3 : 256-col group for PV/output
  const int tileId = wave >> 1;   // 0..3 : S tile (mq,nk)
  const int mq = tileId >> 1, nk = tileId & 1;
  const int half = wave & 1;      // d-half for S reduction

  const v8f vzero = {0.f,0.f,0.f,0.f,0.f,0.f,0.f,0.f};
  v8f acc[16];
  #pragma unroll
  for (int i = 0; i < 16; ++i) acc[i] = vzero;

  const unsigned ksBase = lds_addr(Ks);
  const int kvSteps = blockIdx.x + 1;  // causal: keys 0 .. qBase+31
  for (int step = 0; step < kvSteps; ++step) {
    const int k0 = step * 32;
    __syncthreads();  // LDS reuse fence (Ks/Vt/Pb from previous step)

    // K tile [32][D]: async global->LDS, row-major (already B-friendly)
    {
      const void* gk = (const void*)(Kg + bOff + (size_t)k0 * D);
      for (int i = tid; i < 32 * 128; i += 256) {
        int r = i >> 7, c = (i & 127) * 8;
        async_g2lds_b128(ksBase + (unsigned)(r * KPCH + c) * 2u, gk,
                         (unsigned)(r * D + c) * 2u);
      }
    }
    // V tile transposed into [d][key] via VGPRs (B-operand layout)
    for (int i = tid; i < 32 * 128; i += 256) {
      int r = i >> 7, c = (i & 127) * 8;
      BF8x vv;
      vv.u = *(const uint4*)(Vg + bOff + (size_t)(k0 + r) * D + c);
      #pragma unroll
      for (int j = 0; j < 8; ++j) Vt[(c + j) * VPCH + r] = vv.e[j];
    }
    if (step + 1 < kvSteps) {  // prefetch next tiles into cache hierarchy
      const char* pk = (const char*)(Kg + bOff + (size_t)(k0 + 32) * D);
      const char* pv = (const char*)(Vg + bOff + (size_t)(k0 + 32) * D);
      __builtin_prefetch(pk + tid * 256, 0, 0);
      __builtin_prefetch(pv + tid * 256, 0, 0);
    }
    wait_asynccnt0();   // this wave's async LDS writes (Q on step 0, K always)
    __syncthreads();    // rendezvous: everyone's tiles visible

    // S = Q @ K^T : this wave computes tile (mq,nk) over d-half `half`
    {
      v8f s = vzero;
      const __bf16* aRow = Qs + (size_t)(mq * 16 + lo) * QP;
      const __bf16* bRow = Ks + (size_t)(nk * 16 + lo) * KPCH;
      for (int d0 = half * 512; d0 < half * 512 + 512; d0 += 32) {
        V16 a, b;
        a.h[0] = *(const v8bf*)(aRow + d0 + hi * 8);
        a.h[1] = *(const v8bf*)(aRow + d0 + hi * 8 + 16);
        b.h[0] = *(const v8bf*)(bRow + d0 + hi * 8);
        b.h[1] = *(const v8bf*)(bRow + d0 + hi * 8 + 16);
        s = wmma_bf16(a, b, s);
      }
      #pragma unroll
      for (int r = 0; r < 8; ++r)
        Sb[half * 1024 + (mq * 16 + hi * 8 + r) * 32 + nk * 16 + lo] = s[r];
    }
    __syncthreads();

    // online softmax: one thread per query row (causal mask on key index)
    if (tid < 32) {
      const int row = tid, qRow = qBase + row;
      const float ninf = -__builtin_inff();
      float mOld = rowM[row];
      float mNew = mOld;
      float sv[32];
      #pragma unroll
      for (int j = 0; j < 32; ++j) {
        float x = (k0 + j <= qRow) ? (Sb[row * 32 + j] + Sb[1024 + row * 32 + j])
                                   : ninf;
        sv[j] = x;
        mNew = fmaxf(mNew, x);
      }
      float corr = (mOld == ninf) ? 0.f : __expf(mOld - mNew);
      float lsum = 0.f;
      #pragma unroll
      for (int j = 0; j < 32; ++j) {
        float p = (sv[j] == ninf) ? 0.f : __expf(sv[j] - mNew);
        lsum += p;
        Pb[row * PPCH + j] = (__bf16)p;
      }
      rowM[row] = mNew;
      rowL[row] = rowL[row] * corr + lsum;
      rowC[row] = corr;
    }
    __syncthreads();

    // acc = acc * corr + P @ V   (one WMMA per 16-col tile; K=32 keys)
    {
      V16 a;
      const __bf16* aRow = Pb + (size_t)(mb * 16 + lo) * PPCH;
      a.h[0] = *(const v8bf*)(aRow + hi * 8);
      a.h[1] = *(const v8bf*)(aRow + hi * 8 + 16);
      float cf[8];
      #pragma unroll
      for (int r = 0; r < 8; ++r) cf[r] = rowC[mb * 16 + hi * 8 + r];
      #pragma unroll
      for (int nt = 0; nt < 16; ++nt) {
        const __bf16* bRow = Vt + (size_t)(nb * 256 + nt * 16 + lo) * VPCH + hi * 8;
        V16 b;
        b.h[0] = *(const v8bf*)(bRow);
        b.h[1] = *(const v8bf*)(bRow + 16);
        v8f c = acc[nt];
        #pragma unroll
        for (int r = 0; r < 8; ++r) c[r] *= cf[r];
        acc[nt] = wmma_bf16(a, b, c);
      }
    }
  }
  __syncthreads();

  // epilogue: normalize by l and store f32
  float invL[8];
  #pragma unroll
  for (int r = 0; r < 8; ++r) {
    float l = rowL[mb * 16 + hi * 8 + r];
    invL[r] = (l > 0.f) ? 1.f / l : 0.f;
  }
  for (int nt = 0; nt < 16; ++nt)
    #pragma unroll
    for (int r = 0; r < 8; ++r) {
      int row = qBase + mb * 16 + hi * 8 + r;
      int col = nb * 256 + nt * 16 + lo;
      O[bOff + (size_t)row * D + col] = acc[nt][r] * invL[r];
    }
}

// ---------------------------------------------------------------------------
extern "C" void kernel_launch(void* const* d_in, const int* in_sizes, int n_in,
                              void* d_out, int out_size, void* d_ws, size_t ws_size,
                              hipStream_t stream) {
  (void)in_sizes; (void)n_in; (void)out_size; (void)ws_size;
  const float* Xk = (const float*)d_in[0];
  const float* Xv = (const float*)d_in[1];
  const float* Xq = (const float*)d_in[2];
  const float* Wk = (const float*)d_in[3];
  const float* Wv = (const float*)d_in[4];
  const float* Wq = (const float*)d_in[5];
  float* out = (float*)d_out;

  const size_t elems = (size_t)4 * 2048 * 1024;  // per-tensor bf16 elems
  __bf16* Qw = (__bf16*)d_ws;        // Q (pre-scaled by 1/sqrt(1024))
  __bf16* Kw = Qw + elems;
  __bf16* Vw = Kw + elems;           // total ws use: 3 * 16.78 MB = 50.3 MB

  dim3 gb(8, 64);  // N/128, M/128 (M = 4*2048 = 8192 flattened rows)
  qkv_proj_bf16<<<gb, 256, 0, stream>>>(Xk, Wk, Kw, 1.0f);
  qkv_proj_bf16<<<gb, 256, 0, stream>>>(Xv, Wv, Vw, 1.0f);
  qkv_proj_bf16<<<gb, 256, 0, stream>>>(Xq, Wq, Qw, 0.03125f);  // 1/sqrt(1024)

  // LDS bytes: Qs+Ks (2*32*1032*2) + Vt (1024*40*2) + Sb (8192)
  //          + Pb (2560) + stats (384) = 225152 (< 320KB/WGP)
  const size_t smemBytes = 225152;
  dim3 ga(64, 4);  // 64 query blocks of 32 rows, 4 batches
  flash_attn_causal<<<ga, 256, smemBytes, stream>>>(Qw, Kw, Vw, out);
}
